// _VoxelPointnetSAModuleFSBase_77713138253923
// MI455X (gfx1250) — matrix operand order repaired and checked by reading, not compiled, not run on hardware
//
#include <hip/hip_runtime.h>
#include <hip/hip_bf16.h>

#define B_   4
#define N_   8192
#define CIN  64
#define S_   2048
#define NS   32
#define R2   0.64f
#define K1P  96          // 67 padded to 3 k-steps of 32
#define C1   128
#define C2   256

typedef __attribute__((ext_vector_type(16))) _Float16 v16h;
typedef __attribute__((ext_vector_type(8)))  _Float16 v8h;
typedef __attribute__((ext_vector_type(8)))  float    v8f;

// ---- gfx1250 async-to-LDS path (guarded: falls back to sync loads) --------
#if defined(__has_builtin)
#if __has_builtin(__builtin_amdgcn_global_load_async_to_lds_b32) && \
    __has_builtin(__builtin_amdgcn_s_wait_asynccnt)
#define HAVE_ASYNC_LDS 1
#endif
#endif
#ifndef HAVE_ASYNC_LDS
#define HAVE_ASYNC_LDS 0
#endif

#if HAVE_ASYNC_LDS
typedef __attribute__((address_space(1))) int gint_t;   // global int*
typedef __attribute__((address_space(3))) int lint_t;   // LDS int*
__device__ __forceinline__ void async_copy_dword(const float* gp, float* lp) {
    __builtin_amdgcn_global_load_async_to_lds_b32((gint_t*)gp, (lint_t*)lp, 0, 0);
}
#endif

// ---------------------------------------------------------------------------
// Kernel 1: furthest point sampling, one block per batch, 1024 threads.
// One barrier per iteration: double-buffered reduction slots + redundant
// all-wave final reduction; `last` held in an SGPR via readfirstlane.
// ---------------------------------------------------------------------------
__global__ __launch_bounds__(1024) void fps_kernel(const float* __restrict__ xyz,
                                                   float* __restrict__ newxyz) {
    const int b = blockIdx.x;
    const int t = threadIdx.x;
    const int lane = t & 31;
    const int w = t >> 5;
    const float* X = xyz + (size_t)b * N_ * 3;

    float px[8], py[8], pz[8], dist[8];
#pragma unroll
    for (int i = 0; i < 8; ++i) {
        int p = i * 1024 + t;
        px[i] = X[p * 3 + 0];
        py[i] = X[p * 3 + 1];
        pz[i] = X[p * 3 + 2];
        dist[i] = 1e10f;
    }

    __shared__ float swv[2][32];
    __shared__ int   swi[2][32];

    if (t == 0) {  // first selected index is 0
        newxyz[(b * S_ + 0) * 3 + 0] = X[0];
        newxyz[(b * S_ + 0) * 3 + 1] = X[1];
        newxyz[(b * S_ + 0) * 3 + 2] = X[2];
    }

    int last = 0;
    int par = 0;
    for (int sel = 1; sel < S_; ++sel) {
        float lx = X[last * 3 + 0];   // `last` uniform (SGPR) -> scalar loads
        float ly = X[last * 3 + 1];
        float lz = X[last * 3 + 2];

        float bestv = -1.0f;
        int   besti = N_;
#pragma unroll
        for (int i = 0; i < 8; ++i) {
            float dx = px[i] - lx, dy = py[i] - ly, dz = pz[i] - lz;
            float d = dx * dx + dy * dy + dz * dz;
            dist[i] = fminf(dist[i], d);
            int p = i * 1024 + t;
            if (dist[i] > bestv || (dist[i] == bestv && p < besti)) {
                bestv = dist[i];
                besti = p;
            }
        }
        // wave32 reduction, first-index tie-break
#pragma unroll
        for (int m = 1; m < 32; m <<= 1) {
            float ov = __shfl_xor(bestv, m);
            int   oi = __shfl_xor(besti, m);
            if (ov > bestv || (ov == bestv && oi < besti)) { bestv = ov; besti = oi; }
        }
        if (lane == 0) { swv[par][w] = bestv; swi[par][w] = besti; }
        __syncthreads();
        // every wave redundantly reduces the 32 per-wave winners (no 2nd barrier)
        float v  = swv[par][lane];
        int   ix = swi[par][lane];
#pragma unroll
        for (int m = 1; m < 32; m <<= 1) {
            float ov = __shfl_xor(v, m);
            int   oi = __shfl_xor(ix, m);
            if (ov > v || (ov == v && oi < ix)) { v = ov; ix = oi; }
        }
        last = __builtin_amdgcn_readfirstlane(ix);
        if (t == 0) {
            newxyz[(b * S_ + sel) * 3 + 0] = X[last * 3 + 0];
            newxyz[(b * S_ + sel) * 3 + 1] = X[last * 3 + 1];
            newxyz[(b * S_ + sel) * 3 + 2] = X[last * 3 + 2];
        }
        par ^= 1;   // next iteration writes the other slot buffer
    }
}

// ---------------------------------------------------------------------------
// Kernel 2: ball query, one wave32 per query point.
// ---------------------------------------------------------------------------
__global__ __launch_bounds__(256) void ball_kernel(const float* __restrict__ xyz,
                                                   const float* __restrict__ newxyz,
                                                   int* __restrict__ idxbuf,
                                                   int* __restrict__ cntbuf) {
    const int wid  = (blockIdx.x * blockDim.x + threadIdx.x) >> 5;
    const int lane = threadIdx.x & 31;
    if (wid >= B_ * S_) return;
    const int b = wid / S_;

    const float cx = newxyz[wid * 3 + 0];
    const float cy = newxyz[wid * 3 + 1];
    const float cz = newxyz[wid * 3 + 2];
    const float* X = xyz + (size_t)b * N_ * 3;

    int cnt = 0;
    int firstIdx = -1;
    for (int base = 0; base < N_ && cnt < NS; base += 32) {
        int p = base + lane;
        float dx = X[p * 3 + 0] - cx;
        float dy = X[p * 3 + 1] - cy;
        float dz = X[p * 3 + 2] - cz;
        bool inball = (dx * dx + dy * dy + dz * dz) < R2;
        unsigned mask = (unsigned)__ballot(inball);
        if (inball) {
            int pos = __popc(mask & ((1u << lane) - 1u));
            if (cnt + pos < NS) idxbuf[wid * NS + cnt + pos] = p;
        }
        if (firstIdx < 0 && mask) firstIdx = base + __ffs(mask) - 1;
        cnt += __popc(mask);
    }
    int stored = cnt < NS ? cnt : NS;
    int padval = (cnt > 0) ? firstIdx : 0;
    for (int j = stored + lane; j < NS; j += 32) idxbuf[wid * NS + j] = padval;
    if (lane == 0) cntbuf[wid] = cnt;
}

// ---------------------------------------------------------------------------
// WMMA fragment loaders (CDNA5 16-bit layouts, two 16B LDS loads per fragment).
// ---------------------------------------------------------------------------
__device__ __forceinline__ v16h load_fragA(const _Float16* row, int ks, int hi) {
    union { v16h v; v8h h[2]; } u;
    const _Float16* p = row + ks * 32 + hi * 8;
    u.h[0] = *(const v8h*)p;
    u.h[1] = *(const v8h*)(p + 16);
    return u.v;
}
__device__ __forceinline__ v16h load_fragB(const _Float16* col, int ks, int hi) {
    union { v16h v; v8h h[2]; } u;
    const _Float16* p = col + ks * 32 + hi * 8;
    u.h[0] = *(const v8h*)p;
    u.h[1] = *(const v8h*)(p + 16);
    return u.v;
}
__device__ __forceinline__ v8f wmma_f16(v16h a, v16h b, v8f c) {
    return __builtin_amdgcn_wmma_f32_16x16x32_f16(false, a, false, b, (short)0, c,
                                                  false, false);
}

// ---------------------------------------------------------------------------
// Kernel 3: fused group + MLP(67->128->256) + maxpool, persistent blocks.
// Feature gather for group i+1 streams into LDS via async-to-LDS (ASYNCcnt)
// while group i runs its WMMAs (double-buffered staging).
// ---------------------------------------------------------------------------
#define SM_W1  0
#define SM_W2  (C1 * K1P)                  // 12288
#define SM_G   (SM_W2 + C2 * C1)           // 45056
#define SM_H1  (SM_G + NS * K1P)           // 48128
#define SM_TOT (SM_H1 + NS * C1)           // 52224 halves = 104448 bytes
#define STG_FLOATS (NS * CIN)              // 2048 floats per staging buffer
#define SMEM_BYTES (SM_TOT * 2 + 2 * STG_FLOATS * 4)   // 120832

__global__ __launch_bounds__(256) void mlp_kernel(const float* __restrict__ xyz,
                                                  const float* __restrict__ feats,
                                                  const float* __restrict__ w1,
                                                  const float* __restrict__ b1,
                                                  const float* __restrict__ w2,
                                                  const float* __restrict__ b2,
                                                  const float* __restrict__ newxyz,
                                                  const int* __restrict__ idxbuf,
                                                  const int* __restrict__ cntbuf,
                                                  float* __restrict__ outp) {
    extern __shared__ __align__(16) _Float16 smem[];
    _Float16* w1ld = smem + SM_W1;   // [C1][K1P]
    _Float16* w2ld = smem + SM_W2;   // [C2][C1]
    _Float16* gld  = smem + SM_G;    // [NS][K1P]  (sample-major)
    _Float16* h1ld = smem + SM_H1;   // [NS][C1]   (sample-major)
    float*    stg  = (float*)(smem + SM_TOT);   // [2][NS*CIN] staging

    const int tid  = threadIdx.x;
    const int w    = tid >> 5;
    const int lane = tid & 31;
    const int lm   = lane & 15;
    const int hi   = lane >> 4;
    const int stride = gridDim.x;

    // ---- stage weights (f32 -> f16) into LDS once ----
    for (int e = tid; e < C1 * K1P; e += blockDim.x) {
        int m = e / K1P, k = e % K1P;
        w1ld[e] = (k < CIN + 3) ? (_Float16)w1[m * (CIN + 3) + k] : (_Float16)0.0f;
    }
    for (int e = tid; e < C2 * C1; e += blockDim.x) {
        w2ld[e] = (_Float16)w2[e];
    }
    float bias1[8], bias2[2][8];
#pragma unroll
    for (int r = 0; r < 8; ++r) {
        bias1[r]    = b1[w * 16 + r + 8 * hi];
        bias2[0][r] = b2[(2 * w + 0) * 16 + r + 8 * hi];
        bias2[1][r] = b2[(2 * w + 1) * 16 + r + 8 * hi];
    }

#if HAVE_ASYNC_LDS
    // prologue: async-gather features of the first group into stg[0]
    if (blockIdx.x < B_ * S_) {
        const int grp0 = blockIdx.x;
        const int b0 = grp0 / S_;
        for (int e = tid; e < NS * CIN; e += 256) {
            int n = e >> 6, c = e & 63;
            int idx = idxbuf[grp0 * NS + n];
            async_copy_dword(feats + ((size_t)b0 * CIN + c) * N_ + idx, stg + e);
        }
    }
#endif
    int cur = 0;

    for (int grp = blockIdx.x; grp < B_ * S_; grp += stride) {
        const int b = grp / S_;
        const int s = grp % S_;
        const int cnt = cntbuf[grp];

#if HAVE_ASYNC_LDS
        __builtin_amdgcn_s_wait_asynccnt(0);   // stg[cur] landed
#endif
        __syncthreads();   // prev iter compute done; staging visible to all

        // ---- build g transposed: gld[n][k]; k<3 rel-xyz, 3..66 feat, pad 0
        const float* stgc = stg + cur * STG_FLOATS;
        for (int e = tid; e < NS * K1P; e += blockDim.x) {
            int k = e / NS, n = e % NS;
            float val = 0.0f;
            if (cnt > 0 && k < CIN + 3) {
                if (k < 3) {
                    int idx = idxbuf[grp * NS + n];
                    val = xyz[((size_t)b * N_ + idx) * 3 + k] - newxyz[grp * 3 + k];
                } else {
#if HAVE_ASYNC_LDS
                    val = stgc[n * CIN + (k - 3)];
#else
                    int idx = idxbuf[grp * NS + n];
                    val = feats[((size_t)b * CIN + (k - 3)) * N_ + idx];
#endif
                }
            }
            gld[n * K1P + k] = (_Float16)val;
        }
        __syncthreads();

#if HAVE_ASYNC_LDS
        // ---- kick off next group's gather while this group's WMMAs run
        {
            int nxt = grp + stride;
            if (nxt < B_ * S_) {
                const int bn = nxt / S_;
                float* stgn = stg + (cur ^ 1) * STG_FLOATS;
                for (int e = tid; e < NS * CIN; e += 256) {
                    int n = e >> 6, c = e & 63;
                    int idx = idxbuf[nxt * NS + n];
                    async_copy_dword(feats + ((size_t)bn * CIN + c) * N_ + idx,
                                     stgn + e);
                }
            }
            int pf = grp + 2 * stride;   // prefetch idx rows two groups ahead
            if (pf < B_ * S_ && tid == 0) __builtin_prefetch(idxbuf + pf * NS, 0, 1);
        }
#endif

        // ---- layer 1: wave w owns M-tile w; N-tiles 0,1
        {
            v8f c0, c1;
#pragma unroll
            for (int r = 0; r < 8; ++r) { c0[r] = bias1[r]; c1[r] = bias1[r]; }
            const _Float16* arow = w1ld + (w * 16 + lm) * K1P;
#pragma unroll
            for (int ks = 0; ks < 3; ++ks) {
                v16h a  = load_fragA(arow, ks, hi);
                v16h bA = load_fragB(gld + (0 * 16 + lm) * K1P, ks, hi);
                v16h bB = load_fragB(gld + (1 * 16 + lm) * K1P, ks, hi);
                c0 = wmma_f16(a, bA, c0);
                c1 = wmma_f16(a, bB, c1);
            }
#pragma unroll
            for (int r = 0; r < 8; ++r) {
                int row = w * 16 + r + 8 * hi;
                h1ld[(0 * 16 + lm) * C1 + row] = (_Float16)fmaxf(c0[r], 0.0f);
                h1ld[(1 * 16 + lm) * C1 + row] = (_Float16)fmaxf(c1[r], 0.0f);
            }
        }
        __syncthreads();

        // ---- layer 2: wave w owns M-tiles 2w, 2w+1; N-tiles 0,1
        v8f d[2][2];
#pragma unroll
        for (int mt = 0; mt < 2; ++mt) {
            const int mtile = 2 * w + mt;
#pragma unroll
            for (int r = 0; r < 8; ++r) { d[mt][0][r] = bias2[mt][r]; d[mt][1][r] = bias2[mt][r]; }
            const _Float16* arow = w2ld + (mtile * 16 + lm) * C1;
#pragma unroll
            for (int ks = 0; ks < 4; ++ks) {
                v16h a   = load_fragA(arow, ks, hi);
                v16h b0  = load_fragB(h1ld + (0 * 16 + lm) * C1, ks, hi);
                v16h b1f = load_fragB(h1ld + (1 * 16 + lm) * C1, ks, hi);
                d[mt][0] = wmma_f16(a, b0, d[mt][0]);
                d[mt][1] = wmma_f16(a, b1f, d[mt][1]);
            }
        }

        // ---- relu + maxpool over 32 samples, write pooled (B,256,S)
#pragma unroll
        for (int mt = 0; mt < 2; ++mt) {
#pragma unroll
            for (int r = 0; r < 8; ++r) {
                float m = fmaxf(fmaxf(d[mt][0][r], 0.0f), fmaxf(d[mt][1][r], 0.0f));
                m = fmaxf(m, __shfl_xor(m, 1));
                m = fmaxf(m, __shfl_xor(m, 2));
                m = fmaxf(m, __shfl_xor(m, 4));
                m = fmaxf(m, __shfl_xor(m, 8));
                if (lm == 0) {
                    int row = (2 * w + mt) * 16 + r + 8 * hi;
                    outp[((size_t)b * C2 + row) * S_ + s] = m;
                }
            }
        }
        cur ^= 1;
    }
}

// ---------------------------------------------------------------------------
extern "C" void kernel_launch(void* const* d_in, const int* in_sizes, int n_in,
                              void* d_out, int out_size, void* d_ws, size_t ws_size,
                              hipStream_t stream) {
    const float* xyz   = (const float*)d_in[0];
    const float* feats = (const float*)d_in[1];
    const float* w1    = (const float*)d_in[2];
    const float* b1    = (const float*)d_in[3];
    const float* w2    = (const float*)d_in[4];
    const float* b2    = (const float*)d_in[5];

    float* out    = (float*)d_out;
    float* newxyz = out;                   // (B,S,3)
    float* pooled = out + B_ * S_ * 3;     // (B,256,S)

    int* idxbuf = (int*)d_ws;              // B*S*NS ints
    int* cntbuf = idxbuf + B_ * S_ * NS;   // B*S ints

    fps_kernel<<<B_, 1024, 0, stream>>>(xyz, newxyz);
    ball_kernel<<<(B_ * S_ * 32) / 256, 256, 0, stream>>>(xyz, newxyz, idxbuf, cntbuf);

    (void)hipFuncSetAttribute((const void*)mlp_kernel,
                              hipFuncAttributeMaxDynamicSharedMemorySize,
                              (int)SMEM_BYTES);
    mlp_kernel<<<512, 256, SMEM_BYTES, stream>>>(xyz, feats, w1, b1, w2, b2,
                                                 newxyz, idxbuf, cntbuf, pooled);
}
